// _NonLocalBlockND_120259085110
// MI455X (gfx1250) — matrix operand order repaired and checked by reading, not compiled, not run on hardware
//
#include <hip/hip_runtime.h>

typedef unsigned short u16;
typedef unsigned int   u32;
typedef float   v8f   __attribute__((ext_vector_type(8)));
typedef __bf16  v16bf __attribute__((ext_vector_type(16)));
typedef short   v8s   __attribute__((ext_vector_type(8)));

#define NB    8
#define NC    512
#define ND    256
#define NN    8192
#define NPOOL 2048

static __device__ __forceinline__ u16 f2bf(float f) {
  union { float f; unsigned u; } v; v.f = f;
  unsigned r = v.u + 0x7FFFu + ((v.u >> 16) & 1u);   // round-to-nearest-even
  return (u16)(r >> 16);
}
static __device__ __forceinline__ u32 pack2(float lo, float hi) {
  return (u32)f2bf(lo) | ((u32)f2bf(hi) << 16);
}

union BfPack { v8s s[2]; v16bf v; };
union BfOct  { v8s s;    u16  u[8]; };

// ---- CDNA5 async global->LDS copy (ASYNCcnt) --------------------------------
typedef __attribute__((address_space(3))) char lds_char;
static __device__ __forceinline__ unsigned ldsOff(const void* p) {
  return (unsigned)(unsigned long long)(lds_char*)p;   // generic -> LDS offset
}
static __device__ __forceinline__ void asyncLds16B(unsigned loff, const void* g) {
  asm volatile("global_load_async_to_lds_b128 %0, %1, off"
               :: "v"(loff), "v"(g) : "memory");
}
static __device__ __forceinline__ void asyncWait0() {
  asm volatile("s_wait_asynccnt 0x0" ::: "memory");
}

// A fragment (16x32 bf16, MxK): lane m=L&15, h=L>>4; halves j: K=(j>>3)*16+h*8+(j&7)
static __device__ __forceinline__ v16bf frag_a(const u16* a, int row, int h) {
  BfPack p;
  p.s[0] = *(const v8s*)(a + row * 32 + h * 8);
  p.s[1] = *(const v8s*)(a + row * 32 + 16 + h * 8);
  return p.v;
}
// B fragment (32x16 bf16, KxN): lane n=L&15, h=L>>4; halves j: K=(j>>3)*16+h*8+(j&7)
static __device__ __forceinline__ v16bf frag_b(const u16* bt, int row, int h) {
  BfPack p;
  p.s[0] = *(const v8s*)(bt + row * 32 + h * 8);
  p.s[1] = *(const v8s*)(bt + row * 32 + 16 + h * 8);
  return p.v;
}

// One K=32 step of a 128x128 workgroup tile: 8 waves in 2(M)x4(N) grid,
// each wave owns 4x2 = 8 WMMA 16x16 accumulators.
static __device__ __forceinline__ void mma_8(const u16* sA, const u16* sB,
                                             v8f acc[4][2], int wm, int wn,
                                             int ml, int h) {
  v16bf av[4], bv[2];
#pragma unroll
  for (int i = 0; i < 4; i++) av[i] = frag_a(sA, wm * 64 + i * 16 + ml, h);
#pragma unroll
  for (int j = 0; j < 2; j++) bv[j] = frag_b(sB, wn * 32 + j * 16 + ml, h);
#pragma unroll
  for (int i = 0; i < 4; i++)
#pragma unroll
    for (int j = 0; j < 2; j++)
      acc[i][j] = __builtin_amdgcn_wmma_f32_16x16x32_bf16(
          false, av[i], false, bv[j], (short)0, acc[i][j], false, false);
}

// ---------------- Kernel 1: fused theta/g/phi projections + maxpool ----------
// grid (64 n-tiles, 6 m-tiles over [theta|g|phi], 8 batches)
__global__ __launch_bounds__(256) void k_proj(
    const float* __restrict__ x,
    const float* __restrict__ tw, const float* __restrict__ tb,
    const float* __restrict__ gw, const float* __restrict__ gb,
    const float* __restrict__ pw, const float* __restrict__ pb,
    u16* __restrict__ thetaT, u16* __restrict__ gpool, u16* __restrict__ phipool) {
  __shared__ __align__(16) u16   sA[128 * 32];
  __shared__ __align__(16) u16   sB[128 * 32];
  __shared__ __align__(16) float sC[128 * 64];

  const int b = blockIdx.z;
  const int n0 = blockIdx.x * 128;
  const int wsel = blockIdx.y >> 1;             // 0=theta 1=g 2=phi
  const int mrow0 = (blockIdx.y & 1) * 128;     // row within 256-row weight
  const float* wp = (wsel == 0) ? tw : (wsel == 1) ? gw : pw;
  const float* bp = (wsel == 0) ? tb : (wsel == 1) ? gb : pb;
  const float* xb = x + (size_t)b * NC * NN;

  const int t = threadIdx.x;
  const int lane = t & 31, wid = t >> 5;
  const int wm = wid >> 2, wn = wid & 3;
  const int ml = lane & 15, h = lane >> 4;

  v8f acc[4][2];
  const v8f zf = {0.f, 0.f, 0.f, 0.f, 0.f, 0.f, 0.f, 0.f};
#pragma unroll
  for (int i = 0; i < 4; i++)
#pragma unroll
    for (int j = 0; j < 2; j++) acc[i][j] = zf;

  const int ra = t >> 1, ca = (t & 1) * 16;   // A fill: 2 threads per m row
  const int nB = (t & 31) * 4;                // B fill: 4(K)x4(N) per thread
  const int kB = (t >> 5) * 4;

  for (int kk = 0; kk < NC / 32; ++kk) {
    {  // A tile: 128 m rows x 32 c, f32 -> bf16, packed b128 stores
      const float4* src = (const float4*)(wp + (size_t)(mrow0 + ra) * NC + kk * 32 + ca);
      float4 v0 = src[0], v1 = src[1], v2 = src[2], v3 = src[3];
      uint4 w0, w1;
      w0.x = pack2(v0.x, v0.y); w0.y = pack2(v0.z, v0.w);
      w0.z = pack2(v1.x, v1.y); w0.w = pack2(v1.z, v1.w);
      w1.x = pack2(v2.x, v2.y); w1.y = pack2(v2.z, v2.w);
      w1.z = pack2(v3.x, v3.y); w1.w = pack2(v3.z, v3.w);
      *(uint4*)(sA + ra * 32 + ca)     = w0;
      *(uint4*)(sA + ra * 32 + ca + 8) = w1;
    }
    {  // B tile: x[c, n] -> N-major LDS; pack K-pairs, 4x ds_store_b64
      float q[4][4];
#pragma unroll
      for (int dk = 0; dk < 4; dk++) {
        float4 v = *(const float4*)(xb + (size_t)(kk * 32 + kB + dk) * NN + n0 + nB);
        q[dk][0] = v.x; q[dk][1] = v.y; q[dk][2] = v.z; q[dk][3] = v.w;
      }
#pragma unroll
      for (int dn = 0; dn < 4; dn++) {
        uint2 d;
        d.x = pack2(q[0][dn], q[1][dn]);
        d.y = pack2(q[2][dn], q[3][dn]);
        *(uint2*)(sB + (nB + dn) * 32 + kB) = d;
      }
    }
    if (kk + 1 < NC / 32)
      __builtin_prefetch(xb + (size_t)((kk + 1) * 32 + kB) * NN + n0 + nB, 0, 3);
    __syncthreads();
    mma_8(sA, sB, acc, wm, wn, ml, h);
    __syncthreads();
  }

  if (wsel == 0) {
    // theta: + bias, store transposed (n, d) bf16, 16B per lane store
#pragma unroll
    for (int i = 0; i < 4; i++) {
      const int dbase = mrow0 + wm * 64 + i * 16 + h * 8;
#pragma unroll
      for (int j = 0; j < 2; j++) {
        const int n = n0 + wn * 32 + j * 16 + ml;
        BfOct pk;
#pragma unroll
        for (int r = 0; r < 8; r++) pk.u[r] = f2bf(acc[i][j][r] + bp[dbase + r]);
        *(v8s*)(thetaT + ((size_t)b * NN + n) * ND + dbase) = pk.s;
      }
    }
  } else {
    // g/phi: 2x2 spatial maxpool via LDS staging (pool groups are local to
    // each 64-column chunk because n0 is 128-aligned => y0 multiple of 4)
    u16* outp = (wsel == 1) ? gpool : phipool;
    const int tt = n0 >> 10;              // t index (1024 voxels per t)
    const int y0 = (n0 & 1023) >> 5;      // first y row of tile
#pragma unroll
    for (int chunk = 0; chunk < 2; ++chunk) {
      if ((wn >> 1) == chunk) {
#pragma unroll
        for (int i = 0; i < 4; i++) {
          const int mr = wm * 64 + i * 16 + h * 8;
#pragma unroll
          for (int j = 0; j < 2; j++) {
            const int coll = (wn & 1) * 32 + j * 16 + ml;
#pragma unroll
            for (int r = 0; r < 8; r++) sC[(mr + r) * 64 + coll] = acc[i][j][r];
          }
        }
      }
      __syncthreads();
      {
        const int dl = t >> 1;
        const int px0 = (t & 1) * 8;
        const int d = mrow0 + dl;
        const float bias = bp[d];
        const int py = (y0 >> 1) + chunk;
        const size_t obase = ((size_t)b * ND + d) * NPOOL + tt * 256 + py * 16;
        const float* row = sC + dl * 64;
#pragma unroll
        for (int p = 0; p < 8; p++) {
          const int px = px0 + p;
          float v0 = row[2 * px], v1 = row[2 * px + 1];
          float v2 = row[32 + 2 * px], v3 = row[32 + 2 * px + 1];
          float m = fmaxf(fmaxf(v0, v1), fmaxf(v2, v3));
          outp[obase + px] = f2bf(m + bias);
        }
      }
      __syncthreads();
    }
  }
}

// ---------------- Kernel 2: SgpT[k,d] = sum_m g[d,m] phi[k,m]  (per batch) ---
__global__ __launch_bounds__(256) void k_sgp(const u16* __restrict__ gpool,
                                             const u16* __restrict__ phipool,
                                             float* __restrict__ SgpT) {
  __shared__ __align__(16) u16 sA[128 * 32];
  __shared__ __align__(16) u16 sB[128 * 32];
  const int b = blockIdx.z;
  const int n0 = blockIdx.x * 128;   // phi channel tile
  const int m0 = blockIdx.y * 128;   // g channel tile
  const u16* ga = gpool + (size_t)b * ND * NPOOL;
  const u16* pa = phipool + (size_t)b * ND * NPOOL;
  const int t = threadIdx.x, lane = t & 31, wid = t >> 5;
  const int wm = wid >> 2, wn = wid & 3, ml = lane & 15, h = lane >> 4;
  v8f acc[4][2];
  const v8f zf = {0.f, 0.f, 0.f, 0.f, 0.f, 0.f, 0.f, 0.f};
#pragma unroll
  for (int i = 0; i < 4; i++)
#pragma unroll
    for (int j = 0; j < 2; j++) acc[i][j] = zf;
  const int r2 = t >> 1, c2 = (t & 1) * 16;
  for (int kk = 0; kk < NPOOL / 32; ++kk) {
    const u16* as = ga + (size_t)(m0 + r2) * NPOOL + kk * 32 + c2;
    const u16* bs = pa + (size_t)(n0 + r2) * NPOOL + kk * 32 + c2;
    asyncLds16B(ldsOff(sA + r2 * 32 + c2),     as);
    asyncLds16B(ldsOff(sA + r2 * 32 + c2 + 8), as + 8);
    asyncLds16B(ldsOff(sB + r2 * 32 + c2),     bs);
    asyncLds16B(ldsOff(sB + r2 * 32 + c2 + 8), bs + 8);
    asyncWait0();
    __syncthreads();
    mma_8(sA, sB, acc, wm, wn, ml, h);
    __syncthreads();
  }
  float* outb = SgpT + (size_t)b * ND * ND;
#pragma unroll
  for (int i = 0; i < 4; i++) {
    const int mb = m0 + wm * 64 + i * 16 + h * 8;
#pragma unroll
    for (int j = 0; j < 2; j++) {
      const int n = n0 + wn * 32 + j * 16 + ml;
      *(v8f*)(outb + (size_t)n * ND + mb) = acc[i][j];  // transposed, 32B store
    }
  }
}

// ---------------- Kernel 3: M[c,k] = (1/Np) sum_d w_w[c,d] Sgp[d,k] ----------
__global__ __launch_bounds__(256) void k_mmat(const float* __restrict__ ww,
                                              const float* __restrict__ SgpT,
                                              u16* __restrict__ Mmat) {
  __shared__ __align__(16) u16 sA[128 * 32];
  __shared__ __align__(16) u16 sB[128 * 32];
  const int b = blockIdx.z;
  const int n0 = blockIdx.x * 128;
  const int m0 = blockIdx.y * 128;
  const int t = threadIdx.x, lane = t & 31, wid = t >> 5;
  const int wm = wid >> 2, wn = wid & 3, ml = lane & 15, h = lane >> 4;
  v8f acc[4][2];
  const v8f zf = {0.f, 0.f, 0.f, 0.f, 0.f, 0.f, 0.f, 0.f};
#pragma unroll
  for (int i = 0; i < 4; i++)
#pragma unroll
    for (int j = 0; j < 2; j++) acc[i][j] = zf;
  const int ra = t >> 1, ca = (t & 1) * 16;
  const float* sgb = SgpT + (size_t)b * ND * ND;
  for (int kk = 0; kk < ND / 32; ++kk) {
    {
      const float4* src = (const float4*)(ww + (size_t)(m0 + ra) * ND + kk * 32 + ca);
      float4 v0 = src[0], v1 = src[1], v2 = src[2], v3 = src[3];
      uint4 w0, w1;
      w0.x = pack2(v0.x, v0.y); w0.y = pack2(v0.z, v0.w);
      w0.z = pack2(v1.x, v1.y); w0.w = pack2(v1.z, v1.w);
      w1.x = pack2(v2.x, v2.y); w1.y = pack2(v2.z, v2.w);
      w1.z = pack2(v3.x, v3.y); w1.w = pack2(v3.z, v3.w);
      *(uint4*)(sA + ra * 32 + ca)     = w0;
      *(uint4*)(sA + ra * 32 + ca + 8) = w1;
    }
    {
      const float4* src = (const float4*)(sgb + (size_t)(n0 + ra) * ND + kk * 32 + ca);
      float4 v0 = src[0], v1 = src[1], v2 = src[2], v3 = src[3];
      uint4 w0, w1;
      w0.x = pack2(v0.x, v0.y); w0.y = pack2(v0.z, v0.w);
      w0.z = pack2(v1.x, v1.y); w0.w = pack2(v1.z, v1.w);
      w1.x = pack2(v2.x, v2.y); w1.y = pack2(v2.z, v2.w);
      w1.z = pack2(v3.x, v3.y); w1.w = pack2(v3.z, v3.w);
      *(uint4*)(sB + ra * 32 + ca)     = w0;
      *(uint4*)(sB + ra * 32 + ca + 8) = w1;
    }
    __syncthreads();
    mma_8(sA, sB, acc, wm, wn, ml, h);
    __syncthreads();
  }
  const float inv = 1.0f / (float)NPOOL;
  u16* outb = Mmat + (size_t)b * NC * ND;
#pragma unroll
  for (int i = 0; i < 4; i++) {
    const int mb = m0 + wm * 64 + i * 16 + h * 8;
#pragma unroll
    for (int j = 0; j < 2; j++) {
      const int n = n0 + wn * 32 + j * 16 + ml;
#pragma unroll
      for (int r = 0; r < 8; r++)
        outb[(size_t)(mb + r) * ND + n] = f2bf(acc[i][j][r] * inv);
    }
  }
}

// ---------------- Kernel 4: wy = M @ theta + w_b, accumulate BN stats --------
__global__ __launch_bounds__(256) void k_wy(const u16* __restrict__ Mmat,
                                            const u16* __restrict__ thetaT,
                                            const float* __restrict__ wb,
                                            float* __restrict__ wy,
                                            float* __restrict__ chnSum,
                                            float* __restrict__ chnSq) {
  __shared__ __align__(16) u16 sA[128 * 32];
  __shared__ __align__(16) u16 sB[128 * 32];
  __shared__ float sSum[128];
  __shared__ float sSq[128];
  const int b = blockIdx.z;
  const int n0 = blockIdx.x * 128;
  const int m0 = blockIdx.y * 128;
  const int t = threadIdx.x, lane = t & 31, wid = t >> 5;
  const int wm = wid >> 2, wn = wid & 3, ml = lane & 15, h = lane >> 4;
  if (t < 128) { sSum[t] = 0.f; sSq[t] = 0.f; }
  v8f acc[4][2];
  const v8f zf = {0.f, 0.f, 0.f, 0.f, 0.f, 0.f, 0.f, 0.f};
#pragma unroll
  for (int i = 0; i < 4; i++)
#pragma unroll
    for (int j = 0; j < 2; j++) acc[i][j] = zf;
  const int r2 = t >> 1, c2 = (t & 1) * 16;
  const u16* ma = Mmat + (size_t)b * NC * ND;
  const u16* ta = thetaT + (size_t)b * NN * ND;
  for (int kk = 0; kk < ND / 32; ++kk) {
    const u16* as = ma + (size_t)(m0 + r2) * ND + kk * 32 + c2;
    const u16* bs = ta + (size_t)(n0 + r2) * ND + kk * 32 + c2;
    asyncLds16B(ldsOff(sA + r2 * 32 + c2),     as);
    asyncLds16B(ldsOff(sA + r2 * 32 + c2 + 8), as + 8);
    asyncLds16B(ldsOff(sB + r2 * 32 + c2),     bs);
    asyncLds16B(ldsOff(sB + r2 * 32 + c2 + 8), bs + 8);
    asyncWait0();
    __syncthreads();
    mma_8(sA, sB, acc, wm, wn, ml, h);
    __syncthreads();
  }
  float* wyb = wy + (size_t)b * NC * NN;
#pragma unroll
  for (int i = 0; i < 4; i++) {
    const int mloc = wm * 64 + i * 16 + h * 8;
#pragma unroll
    for (int r = 0; r < 8; r++) {
      const int m = m0 + mloc + r;
      const float bias = wb[m];
      const float v0 = acc[i][0][r] + bias;
      const float v1 = acc[i][1][r] + bias;
      const int na = n0 + wn * 32 + ml;
      wyb[(size_t)m * NN + na]      = v0;
      wyb[(size_t)m * NN + na + 16] = v1;
      atomicAdd(&sSum[mloc + r], v0 + v1);
      atomicAdd(&sSq[mloc + r], v0 * v0 + v1 * v1);
    }
  }
  __syncthreads();
  if (t < 128) {
    atomicAdd(&chnSum[m0 + t], sSum[t]);
    atomicAdd(&chnSq[m0 + t], sSq[t]);
  }
}

// ---------------- Kernel 5: BN (training stats) + residual -------------------
__global__ __launch_bounds__(256) void k_final(const float* __restrict__ wy,
                                               const float* __restrict__ x,
                                               const float* __restrict__ gamma,
                                               const float* __restrict__ beta,
                                               const float* __restrict__ chnSum,
                                               const float* __restrict__ chnSq,
                                               float* __restrict__ out) {
  const size_t i4 = (size_t)blockIdx.x * blockDim.x + threadIdx.x;
  const size_t total4 = (size_t)NB * NC * NN / 4;
  if (i4 >= total4) return;
  const int c = (int)((i4 >> 11) & (NC - 1));  // 2048 float4 per (b,c) row
  const float cnt = (float)(NB * NN);
  const float mean = chnSum[c] / cnt;
  const float var = chnSq[c] / cnt - mean * mean;
  const float scale = gamma[c] * rsqrtf(var + 1e-5f);
  const float shift = beta[c] - mean * scale;
  const float4 w = ((const float4*)wy)[i4];
  const float4 xv = ((const float4*)x)[i4];
  float4 o;
  o.x = w.x * scale + shift + xv.x;
  o.y = w.y * scale + shift + xv.y;
  o.z = w.z * scale + shift + xv.z;
  o.w = w.w * scale + shift + xv.w;
  ((float4*)out)[i4] = o;
}

__global__ void k_zero(float* __restrict__ p, int n) {
  int i = blockIdx.x * blockDim.x + threadIdx.x;
  if (i < n) p[i] = 0.f;
}

extern "C" void kernel_launch(void* const* d_in, const int* in_sizes, int n_in,
                              void* d_out, int out_size, void* d_ws, size_t ws_size,
                              hipStream_t stream) {
  (void)in_sizes; (void)n_in; (void)out_size; (void)ws_size;
  const float* x    = (const float*)d_in[0];
  const float* g_w  = (const float*)d_in[1];
  const float* g_b  = (const float*)d_in[2];
  const float* th_w = (const float*)d_in[3];
  const float* th_b = (const float*)d_in[4];
  const float* ph_w = (const float*)d_in[5];
  const float* ph_b = (const float*)d_in[6];
  const float* w_w  = (const float*)d_in[7];
  const float* w_b  = (const float*)d_in[8];
  const float* bn_g = (const float*)d_in[9];
  const float* bn_b = (const float*)d_in[10];
  float* out = (float*)d_out;

  char* ws = (char*)d_ws;
  u16* thetaT  = (u16*)ws;   ws += (size_t)NB * NN * ND * 2;      // 33.5 MB
  u16* gpool   = (u16*)ws;   ws += (size_t)NB * ND * NPOOL * 2;   //  8.4 MB
  u16* phipool = (u16*)ws;   ws += (size_t)NB * ND * NPOOL * 2;   //  8.4 MB
  float* SgpT  = (float*)ws; ws += (size_t)NB * ND * ND * 4;      //  2.1 MB
  u16* Mmat    = (u16*)ws;   ws += (size_t)NB * NC * ND * 2;      //  2.1 MB
  float* wy    = (float*)ws; ws += (size_t)NB * NC * NN * 4;      //  134 MB
  float* chnSum = (float*)ws; ws += 512 * 4;
  float* chnSq  = (float*)ws;

  k_zero<<<4, 256, 0, stream>>>(chnSum, 1024);
  k_proj<<<dim3(64, 6, 8), 256, 0, stream>>>(x, th_w, th_b, g_w, g_b, ph_w, ph_b,
                                             thetaT, gpool, phipool);
  k_sgp<<<dim3(2, 2, 8), 256, 0, stream>>>(gpool, phipool, SgpT);
  k_mmat<<<dim3(2, 4, 8), 256, 0, stream>>>(w_w, SgpT, Mmat);
  k_wy<<<dim3(64, 4, 8), 256, 0, stream>>>(Mmat, thetaT, w_b, wy, chnSum, chnSq);
  k_final<<<32768, 256, 0, stream>>>(wy, x, bn_g, bn_b, chnSum, chnSq, out);
}
